// OffDiagBlock_38749194944615
// MI455X (gfx1250) — compile-verified, matrix operand order checked by reading, not run on hardware
//
#include <hip/hip_runtime.h>
#include <hip/hip_bf16.h>
#include <math.h>

#define BB   4
#define SIDE 32768
#define DD   256
#define RANK 64
#define NSUP 32
#define GRP  1024
#define KT   (DD / 32)    // 8 k-tiles of 32
#define NT   (RANK / 16)  // 4 n-tiles of 16

typedef __attribute__((ext_vector_type(16))) __bf16 v16bf;
typedef __attribute__((ext_vector_type(8)))  float  v8f;

// ---------------------------------------------------------------------------
// K0: pre-pack top-256 rows of Wu / Wv2 into bf16 WMMA B-fragment layout.
// frag[mat][kt][nt][lane][e], e in 0..15; lane 0-15: n=lane, K = {0..7,16..23}
// (+khalf); lane 16-31: n=lane-16, khalf=8.
// ---------------------------------------------------------------------------
__global__ void k_fragW(const float* __restrict__ Wu, const float* __restrict__ Wv2,
                        __bf16* __restrict__ frag) {
    int idx = blockIdx.x * 256 + threadIdx.x;   // 0..32767
    int e   = idx & 15;
    int l   = (idx >> 4) & 31;
    int nt  = (idx >> 9) & 3;
    int kt  = (idx >> 11) & 7;
    int mat = idx >> 14;
    const float* W = mat ? Wv2 : Wu;            // rows 0..DD-1 (h-part of concat)
    int n     = nt * 16 + (l & 15);
    int khalf = (l >> 4) << 3;
    int k     = kt * 32 + khalf + e + ((e >= 8) ? 8 : 0);
    frag[idx] = (__bf16)W[k * RANK + n];
}

// ---------------------------------------------------------------------------
// K1: segment-mean downsample. One block per (src, b, ns); thread = channel d.
// ---------------------------------------------------------------------------
__global__ void k_downsample(const float* __restrict__ h_row, const float* __restrict__ h_col,
                             float* __restrict__ down_row, float* __restrict__ down_col) {
    int bid = blockIdx.x;             // 0..255
    int sel = bid >> 7;               // 0 = row, 1 = col
    int rem = bid & 127;
    int b   = rem >> 5;
    int ns  = rem & 31;
    const float* src = (sel ? h_col : h_row) +
                       ((size_t)b * SIDE + (size_t)ns * GRP) * DD + threadIdx.x;
    float acc = 0.f;
#pragma unroll 8
    for (int i = 0; i < GRP; ++i) acc += src[(size_t)i * DD];
    float* dst = sel ? down_col : down_row;
    dst[(b * NSUP + ns) * DD + threadIdx.x] = acc * (1.0f / GRP);
}

// ---------------------------------------------------------------------------
// K2a: Q/K/V projections (32x256 @ 256x256, per batch). Block per (sel,b).
// ---------------------------------------------------------------------------
__global__ void k_qkv(const float* __restrict__ down_row, const float* __restrict__ down_col,
                      const float* __restrict__ Wq, const float* __restrict__ bq,
                      const float* __restrict__ Wk, const float* __restrict__ bk,
                      const float* __restrict__ Wv, const float* __restrict__ bv,
                      float* __restrict__ Qg, float* __restrict__ Kg, float* __restrict__ Vg) {
    int sel = blockIdx.x >> 2;        // 0=Q 1=K 2=V
    int b   = blockIdx.x & 3;
    int d   = threadIdx.x;
    const float* down = (sel == 0 ? down_row : down_col) + (size_t)(b * NSUP) * DD;
    const float* W    = (sel == 0) ? Wq : ((sel == 1) ? Wk : Wv);
    const float* bias = (sel == 0) ? bq : ((sel == 1) ? bk : bv);
    float* out = ((sel == 0) ? Qg : ((sel == 1) ? Kg : Vg)) + (size_t)(b * NSUP) * DD;
    for (int n = 0; n < NSUP; ++n) {
        float acc = bias[d];
#pragma unroll 4
        for (int k = 0; k < DD; ++k) acc += down[n * DD + k] * W[k * DD + d];
        out[n * DD + d] = acc;
    }
}

// ---------------------------------------------------------------------------
// K2b: 32x32 masked softmax attention, block per batch.
// ---------------------------------------------------------------------------
__global__ void k_attn(const float* __restrict__ Qg, const float* __restrict__ Kg,
                       const float* __restrict__ Vg, const unsigned char* __restrict__ mask,
                       float* __restrict__ row_out) {
    __shared__ float ssc[NSUP][NSUP];
    int b = blockIdx.x;
    int t = threadIdx.x;
    const float sa = 0.0625f;  // 256^-0.5
    for (int idx = t; idx < NSUP * NSUP; idx += 256) {
        int n = idx >> 5, m = idx & 31;
        float acc = 0.f;
#pragma unroll 4
        for (int k = 0; k < DD; ++k)
            acc += Qg[(size_t)(b * NSUP + n) * DD + k] * Kg[(size_t)(b * NSUP + m) * DD + k];
        acc *= sa;
        if (!mask[idx]) acc = -__builtin_inff();
        ssc[n][m] = acc;
    }
    __syncthreads();
    if (t < NSUP) {
        float mx = -__builtin_inff();
        for (int m = 0; m < NSUP; ++m) mx = fmaxf(mx, ssc[t][m]);
        float sum = 0.f;
        for (int m = 0; m < NSUP; ++m) { float e = __expf(ssc[t][m] - mx); ssc[t][m] = e; sum += e; }
        float inv = 1.f / sum;
        for (int m = 0; m < NSUP; ++m) ssc[t][m] *= inv;
    }
    __syncthreads();
    int d = t;
    for (int n = 0; n < NSUP; ++n) {
        float acc = 0.f;
#pragma unroll
        for (int m = 0; m < NSUP; ++m)
            acc += ssc[n][m] * Vg[(size_t)(b * NSUP + m) * DD + d];
        row_out[(size_t)(b * NSUP + n) * DD + d] = acc;
    }
}

// ---------------------------------------------------------------------------
// K2c: fold group-constant term + bias:
//   ru[b,ns,:] = bu + row_out[b,ns,:] @ Wu[256:512,:]
//   cv[b,ns,:] = bv2 + down_col[b,ns,:] @ Wv2[256:512,:]
// ---------------------------------------------------------------------------
__global__ void k_fold(const float* __restrict__ row_out, const float* __restrict__ down_col,
                       const float* __restrict__ Wu, const float* __restrict__ bu,
                       const float* __restrict__ Wv2, const float* __restrict__ bv2,
                       float* __restrict__ ru, float* __restrict__ cv) {
    int mat = blockIdx.x >> 2;
    int b   = blockIdx.x & 3;
    const float* X  = mat ? down_col : row_out;
    const float* W2 = (mat ? Wv2 : Wu) + (size_t)DD * RANK;   // rows DD..2DD-1
    const float* bb = mat ? bv2 : bu;
    float* out = mat ? cv : ru;
    for (int i = threadIdx.x; i < NSUP * RANK; i += 256) {
        int ns = i >> 6, r = i & 63;
        float acc = bb[r];
#pragma unroll 4
        for (int k = 0; k < DD; ++k)
            acc += X[(size_t)(b * NSUP + ns) * DD + k] * W2[k * RANK + r];
        out[(b * NSUP + ns) * RANK + r] = acc;
    }
}

// ---------------------------------------------------------------------------
// K3: hot GEMM. One wave per 16x64 output strip of U (mat=0) or Vt (mat=1):
//   out[row, :] = (h[row, :256] @ Wfrag + fold[b,ns,:]) * scale
// 8 K-steps x 4 N-tiles of v_wmma_f32_16x16x32_bf16, f32 accumulate.
// ---------------------------------------------------------------------------
__global__ void __launch_bounds__(256) k_bigmm(
    const float* __restrict__ h_row, const float* __restrict__ h_col,
    const __bf16* __restrict__ frag,
    const float* __restrict__ ru, const float* __restrict__ cv,
    const float* __restrict__ scale,
    float* __restrict__ out /* U then Vt, flat */) {
    int wave  = blockIdx.x * 8 + (threadIdx.x >> 5);   // 0..16383
    int lane  = threadIdx.x & 31;
    int mat   = wave >> 13;                            // 8192 waves per matrix
    int rem   = wave & 8191;
    int b     = rem >> 11;
    int mtile = rem & 2047;
    int row0  = mtile << 4;
    int ns    = row0 >> 10;

    const float* h    = (mat ? h_col : h_row) + (size_t)b * SIDE * DD;
    const float* addv = (mat ? cv : ru) + (size_t)(b * NSUP + ns) * RANK;
    float* o = out + (size_t)mat * BB * SIDE * RANK + (size_t)b * SIDE * RANK;
    const __bf16* fb = frag + (size_t)mat * (KT * NT * 32 * 16);

    int mrow  = lane & 15;
    int khalf = (lane >> 4) << 3;                      // 0 or 8
    const float* hr = h + (size_t)(row0 + mrow) * DD;

    v8f c[NT];
#pragma unroll
    for (int i = 0; i < NT; ++i) {
        v8f z = {0.f, 0.f, 0.f, 0.f, 0.f, 0.f, 0.f, 0.f};
        c[i] = z;
    }

#pragma unroll
    for (int kt = 0; kt < KT; ++kt) {
        const float* p = hr + kt * 32 + khalf;
        float4 x0 = *(const float4*)(p);        // K = khalf+0..3
        float4 x1 = *(const float4*)(p + 4);    // K = khalf+4..7
        float4 x2 = *(const float4*)(p + 16);   // K = 16+khalf+0..3
        float4 x3 = *(const float4*)(p + 20);   // K = 16+khalf+4..7
        v16bf a;
        a[0]  = (__bf16)x0.x; a[1]  = (__bf16)x0.y; a[2]  = (__bf16)x0.z; a[3]  = (__bf16)x0.w;
        a[4]  = (__bf16)x1.x; a[5]  = (__bf16)x1.y; a[6]  = (__bf16)x1.z; a[7]  = (__bf16)x1.w;
        a[8]  = (__bf16)x2.x; a[9]  = (__bf16)x2.y; a[10] = (__bf16)x2.z; a[11] = (__bf16)x2.w;
        a[12] = (__bf16)x3.x; a[13] = (__bf16)x3.y; a[14] = (__bf16)x3.z; a[15] = (__bf16)x3.w;
#pragma unroll
        for (int nt = 0; nt < NT; ++nt) {
            v16bf bf = *(const v16bf*)(fb + (size_t)((kt * NT + nt) * 32 + lane) * 16);
            c[nt] = __builtin_amdgcn_wmma_f32_16x16x32_bf16(
                false, a, false, bf, (short)0, c[nt], false, false);
        }
    }

    float s     = scale[0];
    int   mbase = row0 + ((lane >> 4) << 3);
#pragma unroll
    for (int nt = 0; nt < NT; ++nt) {
        int   n   = nt * 16 + (lane & 15);
        float add = addv[n];
#pragma unroll
        for (int r = 0; r < 8; ++r)
            o[(size_t)(mbase + r) * RANK + n] = (c[nt][r] + add) * s;
    }
}

// ---------------------------------------------------------------------------
extern "C" void kernel_launch(void* const* d_in, const int* in_sizes, int n_in,
                              void* d_out, int out_size, void* d_ws, size_t ws_size,
                              hipStream_t stream) {
    const float* h_row = (const float*)d_in[0];
    const float* h_col = (const float*)d_in[1];
    const unsigned char* mask = (const unsigned char*)d_in[2];
    const float* scale = (const float*)d_in[3];
    const float* Wq  = (const float*)d_in[4];  const float* bq  = (const float*)d_in[5];
    const float* Wk  = (const float*)d_in[6];  const float* bk  = (const float*)d_in[7];
    const float* Wv  = (const float*)d_in[8];  const float* bv  = (const float*)d_in[9];
    const float* Wu  = (const float*)d_in[10]; const float* bu  = (const float*)d_in[11];
    const float* Wv2 = (const float*)d_in[12]; const float* bv2 = (const float*)d_in[13];

    float* f = (float*)d_ws;
    float* down_row = f;                // B*NS*D = 32768
    float* down_col = f + 32768;
    float* Qg       = f + 65536;
    float* Kg       = f + 98304;
    float* Vg       = f + 131072;
    float* row_out  = f + 163840;
    float* ru       = f + 196608;       // B*NS*RANK = 8192
    float* cv       = f + 204800;
    __bf16* frag    = (__bf16*)(f + 212992);  // 2*8*4*32*16 bf16 = 64 KB

    k_fragW     <<<128,  256, 0, stream>>>(Wu, Wv2, frag);
    k_downsample<<<256,  256, 0, stream>>>(h_row, h_col, down_row, down_col);
    k_qkv       <<<12,   256, 0, stream>>>(down_row, down_col, Wq, bq, Wk, bk, Wv, bv, Qg, Kg, Vg);
    k_attn      <<<4,    256, 0, stream>>>(Qg, Kg, Vg, mask, row_out);
    k_fold      <<<8,    256, 0, stream>>>(row_out, down_col, Wu, bu, Wv2, bv2, ru, cv);
    k_bigmm     <<<2048, 256, 0, stream>>>(h_row, h_col, frag, ru, cv, scale, (float*)d_out);

    (void)in_sizes; (void)n_in; (void)out_size; (void)ws_size;
}
